// PhotonicDelayReservoir_88295937671438
// MI455X (gfx1250) — matrix operand order repaired and checked by reading, not compiled, not run on hardware
//
#include <hip/hip_runtime.h>
#include <math.h>

// ---------------------------------------------------------------------------
// PhotonicDelayReservoir on MI455X (gfx1250, wave32, WMMA)
//
// 16 persistent workgroups x 256 threads (8 waves). WG g owns output columns
// [32g, 32g+32). Its W_fb slice (5 taps x 512 K x 32 cols, bf16) lives in LDS
// in WMMA-B fragment layout for the whole kernel. Delayed states live in a
// 256-slot bf16 ring in global scratch (L2-resident, ~8.4 MB). Each timestep:
//   waves compute k-split partial WMMA accumulations (v_wmma_f32_16x16x32_bf16)
//   across 4 independent accumulator chains (latency hiding), LDS tree-reduce,
//   tanh/leak epilogue, write f32 out + f32 hstate + bf16 ring, then a 16-WG
//   atomic barrier (release/acquire at agent scope).
// ---------------------------------------------------------------------------

typedef __bf16 bf16;
typedef __attribute__((ext_vector_type(16))) __bf16        v16bf;
typedef __attribute__((ext_vector_type(8)))  float         v8f;
typedef __attribute__((ext_vector_type(4)))  unsigned int  u32x4;

#define B_        32
#define S_        4096
#define DIN_      8
#define R_        512
#define NTAPS_    5
#define NWG_      16
#define NCOL_     32                 // output columns per workgroup
#define RING_SLOTS_ 256              // power of 2 > MAX_DELAY(168): avoids WAR race
#define RING_STRIDE_ (B_ * R_)       // 16384 bf16 per slot
#define LEAK_     0.1f

// LDS carve (dynamic shared):
//   wfb : 5*16*2 fragments * 1KB                = 163840 B
//   red : 8 waves * 2 accs * 256 f32            =  16384 B
//   xs  : 32*8 f32                              =   1024 B
//   wi  : 32*8 f32                              =   1024 B
//   bs  : 32 f32 (+pad)                         =    128 B
#define SMEM_WFB_   0
#define SMEM_RED_   163840
#define SMEM_XS_    (163840 + 16384)
#define SMEM_WI_    (163840 + 16384 + 1024)
#define SMEM_BS_    (163840 + 16384 + 2048)
#define SMEM_TOTAL_ (163840 + 16384 + 2048 + 128)

__global__ void pdr_init_kernel(unsigned* cnt) {
    if (threadIdx.x == 0) *cnt = 0u;
}

__global__ void __launch_bounds__(256, 1)
pdr_reservoir_kernel(const float* __restrict__ xg,     // (B, S, DIN)
                     const float* __restrict__ Win,    // (R, DIN)
                     const float* __restrict__ Wfb,    // (NTAPS, R, R)
                     const float* __restrict__ bias,   // (R)
                     float*       __restrict__ out,    // (B, S, R)
                     unsigned*    __restrict__ cnt,    // grid barrier counter
                     bf16*        __restrict__ ring,   // (RING_SLOTS, B, R)
                     float*       __restrict__ hstate) // (B, R) f32 leak state
{
    constexpr int taps[NTAPS_] = {1, 4, 24, 96, 168};

    extern __shared__ char smem[];
    bf16*  wfb = (bf16*)(smem + SMEM_WFB_);
    float* red = (float*)(smem + SMEM_RED_);
    float* xs  = (float*)(smem + SMEM_XS_);
    float* wi  = (float*)(smem + SMEM_WI_);
    float* bs  = (float*)(smem + SMEM_BS_);

    const int tid  = (int)threadIdx.x;
    const int lane = tid & 31;
    const int w    = tid >> 5;       // wave id 0..7
    const int mt   = w >> 2;         // M-tile owned by this wave (batch 16-group)
    const int ksp  = w & 3;          // k-split lane of this wave
    const int wg   = (int)blockIdx.x;
    const int colbase = wg * NCOL_;

    // ---- one-time prefill: W_fb slice -> LDS in WMMA B-fragment layout ----
    // fragment f = (tap*16 + kb)*2 + nt : a 32x16 bf16 block; lane L stores its
    // 16 values contiguously (32 B) so the WMMA read is a single v16bf load.
// In-fragment value jj of lane L:  K = kb*32 + (jj>=8 ? 16 : 0) + (L>=16 ? 8 : 0) + (jj & 7),
    //                                  N = colbase + nt*16 + (L & 15).
    for (int v = tid; v < NTAPS_ * 16 * 2 * 512; v += 256) {
        int f   = v >> 9;
        int q   = v & 511;
        int ln  = q >> 4;
        int jj  = q & 15;
        int tap = f / 32;
        int rem = f - tap * 32;
        int kb  = rem >> 1;
        int nt  = rem & 1;
        int k   = kb * 32 + ((jj & 8) ? 16 : 0) + ((ln & 16) ? 8 : 0) + (jj & 7);
        int col = colbase + nt * 16 + (ln & 15);
        wfb[f * 512 + ln * 16 + jj] = (bf16)Wfb[(tap * R_ + k) * R_ + col];
    }
    {   // W_in slice and bias slice
        int cl = tid >> 3, i = tid & 7;
        wi[cl * 8 + i] = Win[(colbase + cl) * DIN_ + i];
        if (i == 0) bs[cl] = bias[colbase + cl];
    }
    __syncthreads();

    const int arow = mt * 16 + (lane & 15);      // A-matrix row (batch index)
    const int koff = (lane & 16) ? 8 : 0;        // A-fragment K half-offset

    for (int t = 0; t < S_; ++t) {
        // stage x_t (32 x 8 f32) into LDS; consumed after the reduce barrier
        {
            int b = tid >> 3, i = tid & 7;
            xs[tid] = xg[b * (S_ * DIN_) + t * DIN_ + i];
        }

        // 4 independent accumulator chains: (q parity) x (N-tile).
        v8f accA0 = {}, accA1 = {};   // q even
        v8f accB0 = {}, accB1 = {};   // q odd
#pragma unroll
        for (int tap = 0; tap < NTAPS_; ++tap) {
            if (t >= taps[tap]) {   // skip => implicit zero history (deterministic)
                const bf16* slotp = ring
                    + (size_t)((unsigned)(t - taps[tap]) & (RING_SLOTS_ - 1)) * RING_STRIDE_
                    + arow * R_ + koff;
#pragma unroll
                for (int q = 0; q < 4; ++q) {
                    const int kbl = ksp + 4 * q;                 // k-block 0..15
                    const bf16* rp = slotp + kbl * 32;
                    union { v16bf v; struct { u32x4 lo, hi; } p; } a;
                    a.p.lo = *(const u32x4*)(rp);                // K .. K+7
                    a.p.hi = *(const u32x4*)(rp + 16);           // K+16 .. K+23
                    const int f0 = (tap * 16 + kbl) * 2;
                    v16bf b0 = *(const v16bf*)(wfb + f0 * 512 + lane * 16);
                    v16bf b1 = *(const v16bf*)(wfb + f0 * 512 + 512 + lane * 16);
                    if (q & 1) {
                        accB0 = __builtin_amdgcn_wmma_f32_16x16x32_bf16(
                            false, a.v, false, b0, (short)0, accB0, false, false);
                        accB1 = __builtin_amdgcn_wmma_f32_16x16x32_bf16(
                            false, a.v, false, b1, (short)0, accB1, false, false);
                    } else {
                        accA0 = __builtin_amdgcn_wmma_f32_16x16x32_bf16(
                            false, a.v, false, b0, (short)0, accA0, false, false);
                        accA1 = __builtin_amdgcn_wmma_f32_16x16x32_bf16(
                            false, a.v, false, b1, (short)0, accA1, false, false);
                    }
                }
            }
        }
        v8f acc0 = accA0 + accB0;
        v8f acc1 = accA1 + accB1;

        // ---- k-split reduction through LDS ----
#pragma unroll
        for (int r = 0; r < 8; ++r) {
            red[(w * 2 + 0) * 256 + r * 32 + lane] = acc0[r];
            red[(w * 2 + 1) * 256 + r * 32 + lane] = acc1[r];
        }
        __syncthreads();

        if (w < 4) {                       // 4 finishing waves: one 16x16 tile each
            const int fmt = w >> 1;        // M-tile
            const int fnt = w & 1;         // N-tile
#pragma unroll
            for (int r = 0; r < 8; ++r) {
                float s = 0.f;
#pragma unroll
                for (int jj = 0; jj < 4; ++jj)
                    s += red[((fmt * 4 + jj) * 2 + fnt) * 256 + r * 32 + lane];
                const int M   = r + ((lane & 16) ? 8 : 0);
                const int n   = lane & 15;
                const int b   = fmt * 16 + M;
                const int cl  = fnt * 16 + n;
                const int col = colbase + cl;
                float drive = 0.f;
#pragma unroll
                for (int i = 0; i < DIN_; ++i)
                    drive = fmaf(xs[b * 8 + i], wi[cl * 8 + i], drive);
                const float pre = drive + s + bs[cl];
                const float hp  = (t > 0) ? hstate[b * R_ + col] : 0.f;
                const float hn  = (1.0f - LEAK_) * hp + LEAK_ * tanhf(pre);
                hstate[b * R_ + col] = hn;
                out[(size_t)(b * S_ + t) * R_ + col] = hn;
                ring[(size_t)(t & (RING_SLOTS_ - 1)) * RING_STRIDE_ + b * R_ + col] = (bf16)hn;
            }
        }

        // ---- grid barrier across the 16 persistent workgroups ----
        __threadfence();                   // drain this thread's global stores
        __syncthreads();
        if (tid == 0) {
            __hip_atomic_fetch_add(cnt, 1u, __ATOMIC_RELEASE, __HIP_MEMORY_SCOPE_AGENT);
            const unsigned target = (unsigned)NWG_ * (unsigned)(t + 1);
            while (__hip_atomic_load(cnt, __ATOMIC_RELAXED, __HIP_MEMORY_SCOPE_AGENT) < target)
                __builtin_amdgcn_s_sleep(1);
        }
        __syncthreads();
        __builtin_amdgcn_fence(__ATOMIC_ACQUIRE, "agent");  // invalidate: see ring
    }
}

extern "C" void kernel_launch(void* const* d_in, const int* in_sizes, int n_in,
                              void* d_out, int out_size, void* d_ws, size_t ws_size,
                              hipStream_t stream) {
    (void)in_sizes; (void)n_in; (void)out_size; (void)ws_size;
    const float* xg   = (const float*)d_in[0];   // (32, 4096, 8)
    const float* Win  = (const float*)d_in[1];   // (512, 8)
    const float* Wfb  = (const float*)d_in[2];   // (5, 512, 512)
    const float* bias = (const float*)d_in[3];   // (512)
    float* out = (float*)d_out;                  // (32, 4096, 512)

    char* ws = (char*)d_ws;
    unsigned* cnt  = (unsigned*)ws;                                   // 256 B slot
    bf16*     ring = (bf16*)(ws + 256);                               // ~8.4 MB
    float*    hst  = (float*)(ws + 256 +
                              (size_t)RING_SLOTS_ * RING_STRIDE_ * sizeof(bf16));

    pdr_init_kernel<<<1, 32, 0, stream>>>(cnt);
    pdr_reservoir_kernel<<<NWG_, 256, SMEM_TOTAL_, stream>>>(
        xg, Win, Wfb, bias, out, cnt, ring, hst);
}
